// GraphAttentionLayer_59021440582054
// MI455X (gfx1250) — matrix operand order repaired and checked by reading, not compiled, not run on hardware
//
#include <hip/hip_runtime.h>

// ---- types matching gfx1250 WMMA operand layouts ----
typedef _Float16 v16h __attribute__((ext_vector_type(16)));
typedef _Float16 h8   __attribute__((ext_vector_type(8)));
typedef float    v8f  __attribute__((ext_vector_type(8)));
typedef float    v4f  __attribute__((ext_vector_type(4)));
typedef float    v2f  __attribute__((ext_vector_type(2)));
typedef int      v4i  __attribute__((ext_vector_type(4)));

#define N_NODES 8192
#define FDIM    256
#define ALPHA   0.2f

// =====================================================================
// Kernel 1: h = input @ W  (fp32), also emit hT (f16, [F][N]) for the
// attention GEMM's B operands. Exact-precision f32 WMMA path.
// Block = 512 thr = 16 waves; wave w -> n-tile w; block -> 16-row m-tile.
// =====================================================================
__global__ __launch_bounds__(512)
void gat_h_gemm(const float* __restrict__ input, const float* __restrict__ W,
                float* __restrict__ h_out, _Float16* __restrict__ hT) {
  const int tid   = threadIdx.x;
  const int wave  = tid >> 5;
  const int lane  = tid & 31;
  const int halfw = lane >> 4;   // lane group 0/1
  const int l     = lane & 15;
  const int m0    = blockIdx.x * 16;
  const int n0    = wave * 16;

  v8f acc = {};
#if __has_builtin(__builtin_amdgcn_wmma_f32_16x16x4_f32)
  for (int k0 = 0; k0 < FDIM; k0 += 4) {
    // A 16x4 f32: VGPR v -> K = halfw*2 + v, M = l   (ISA 7.12.2)
    const float* ap = input + (size_t)(m0 + l) * FDIM + k0 + halfw * 2;
    v2f a; a[0] = ap[0]; a[1] = ap[1];
    // B 4x16 f32: lanes0-15 K=0..1, lanes16-31 K=2..3, N = l
    v2f b;
    b[0] = W[(size_t)(k0 + halfw * 2 + 0) * FDIM + n0 + l];
    b[1] = W[(size_t)(k0 + halfw * 2 + 1) * FDIM + n0 + l];
    acc = __builtin_amdgcn_wmma_f32_16x16x4_f32(false, a, false, b,
                                                (short)0, acc, false, false);
  }
#else
  for (int k = 0; k < FDIM; ++k) {
    float bv = W[(size_t)k * FDIM + n0 + l];
#pragma unroll
    for (int g = 0; g < 8; ++g)
      acc[g] += input[(size_t)(m0 + halfw * 8 + g) * FDIM + k] * bv;
  }
#endif
  // C/D 16x16 f32 layout: VGPR g -> M = halfw*8 + g, N = l
#pragma unroll
  for (int g = 0; g < 8; ++g) {
    int m = m0 + halfw * 8 + g;
    int n = n0 + l;
    float v = acc[g];
    h_out[(size_t)m * FDIM + n] = v;
    hT[(size_t)n * N_NODES + m] = (_Float16)v;
  }
}

// =====================================================================
// Kernel 2: src[i] = h[i]·a[0:F], dst[i] = h[i]·a[F:2F].  Wave per row.
// =====================================================================
__global__ __launch_bounds__(256)
void gat_src_dst(const float* __restrict__ h, const float* __restrict__ avec,
                 float* __restrict__ src, float* __restrict__ dst) {
  const int wave = threadIdx.x >> 5;
  const int lane = threadIdx.x & 31;
  const int row  = blockIdx.x * 8 + wave;
  float ss = 0.f, dd = 0.f;
#pragma unroll
  for (int q = 0; q < 2; ++q) {
    int idx = lane * 4 + q * 128;
    v4f hv = *(const v4f*)(h + (size_t)row * FDIM + idx);
    v4f as = *(const v4f*)(avec + idx);
    v4f ad = *(const v4f*)(avec + FDIM + idx);
    ss += hv[0]*as[0] + hv[1]*as[1] + hv[2]*as[2] + hv[3]*as[3];
    dd += hv[0]*ad[0] + hv[1]*ad[1] + hv[2]*ad[2] + hv[3]*ad[3];
  }
  for (int off = 16; off > 0; off >>= 1) {
    ss += __shfl_xor(ss, off, 32);
    dd += __shfl_xor(dd, off, 32);
  }
  if (lane == 0) { src[row] = ss; dst[row] = dd; }
}

// =====================================================================
// Kernel 3: max over dst -> safe (exact, shift-invariant) softmax shift.
// =====================================================================
__global__ __launch_bounds__(256)
void gat_maxdst(const float* __restrict__ dst, float* __restrict__ maxdst) {
  __shared__ float red[256];
  float m = -3.4e38f;
  for (int j = threadIdx.x; j < N_NODES; j += 256) m = fmaxf(m, dst[j]);
  red[threadIdx.x] = m;
  __syncthreads();
  for (int s = 128; s > 0; s >>= 1) {
    if (threadIdx.x < s) red[threadIdx.x] = fmaxf(red[threadIdx.x], red[threadIdx.x + s]);
    __syncthreads();
  }
  if (threadIdx.x == 0) maxdst[0] = red[0];
}

// =====================================================================
// Kernel 4: fused attention. Block = 256 thr (8 waves) owns 16 rows.
// Stream adj in 16x128 tiles (2x b128 NT loads/thread), compute
// p = exp(leaky(src+dst)-m) (0 where masked), stage P (f16) in a
// double-buffered, bank-conflict-padded LDS tile, and accumulate
// P @ hT with v_wmma_f32_16x16x32_f16. One barrier per tile.
// Epilogue: out = relu(acc / rowsum).
// =====================================================================
#define JT  128        // j columns per tile
#define STR 136        // padded LDS row stride in halves (272B = 68 dwords)

__global__ __launch_bounds__(256)
void gat_attn(const int* __restrict__ adj, const _Float16* __restrict__ hT,
              const float* __restrict__ src, const float* __restrict__ dst,
              const float* __restrict__ maxdst, float* __restrict__ out) {
  __shared__ __align__(16) _Float16 P[2][16 * STR];  // double-buffered p tile
  __shared__ float red[256];
  __shared__ float sS[16];     // src per row
  __shared__ float sM[16];     // softmax shift per row
  __shared__ float sLinv[16];  // 1 / rowsum

  const int tid   = threadIdx.x;
  const int wave  = tid >> 5;
  const int lane  = tid & 31;
  const int halfw = lane >> 4;
  const int l     = lane & 15;
  const int r0    = blockIdx.x * 16;
  const int n0    = wave * 32;          // 2 n-tiles per wave -> F=256 over 8 waves

  // producer mapping: 256 thr cover 16 rows x 128 cols, 8 consecutive cols/thread
  const int pr = tid >> 4;
  const int pc = (tid & 15) * 8;

  if (tid < 16) {
    float s = src[r0 + tid];
    float x = s + maxdst[0];
    sS[tid] = s;
    sM[tid] = x > 0.f ? x : ALPHA * x;   // leaky is monotone -> valid shift
  }
  __syncthreads();
  const float s_r = sS[pr];
  const float m_r = sM[pr];

  const int*   aptr = adj + (size_t)(r0 + pr) * N_NODES + pc;
  const float* dptr = dst + pc;

  v8f acc0 = {};
  v8f acc1 = {};
  float psum = 0.f;

  for (int jt = 0; jt < N_NODES / JT; ++jt) {
    const int j0 = jt * JT;
    _Float16* Pb = &P[jt & 1][0];

    // adj read exactly once device-wide: non-temporal, keep L2 for hT
    v4i av0 = __builtin_nontemporal_load((const v4i*)(aptr + j0));
    v4i av1 = __builtin_nontemporal_load((const v4i*)(aptr + j0 + 4));
    __builtin_prefetch(aptr + j0 + JT, 0, 1);   // global_prefetch next tile
    v4f dv0 = *(const v4f*)(dptr + j0);
    v4f dv1 = *(const v4f*)(dptr + j0 + 4);

    h8 pv;
#pragma unroll
    for (int q = 0; q < 8; ++q) {
      int   msk = (q < 4) ? av0[q]     : av1[q - 4];
      float d   = (q < 4) ? dv0[q]     : dv1[q - 4];
      float x = s_r + d;
      float e = x > 0.f ? x : ALPHA * x;
      float p = (msk > 0) ? __expf(e - m_r) : 0.f;  // mask == exp(-9e15-m) == 0
      psum += p;
      pv[q] = (_Float16)p;
    }
    *(h8*)&Pb[pr * STR + pc] = pv;   // b128 store, padded stride -> conflict-free
    __syncthreads();                 // single barrier/tile (double buffer covers WAR)

    // consume: 4 K-blocks of 32, 2 n-tiles -> 8 WMMA per wave per tile
#pragma unroll
    for (int kb = 0; kb < JT; kb += 32) {
      // A 16x32 f16 (ISA 7.12.2): elem e -> K = kb + (e>=8?16:0) + halfw*8 + (e&7), M=l
      h8 a0 = *(const h8*)&Pb[l * STR + kb + halfw * 8];
      h8 a1 = *(const h8*)&Pb[l * STR + kb + 16 + halfw * 8];
      v16h Am = __builtin_shufflevector(a0, a1, 0,1,2,3,4,5,6,7,8,9,10,11,12,13,14,15);
#pragma unroll
      for (int nt = 0; nt < 2; ++nt) {
        // B 32x16 f16: elem e -> K = (e>=8?16:0) + halfw*8 + (e&7), N=l; hT row-contiguous in K
        const _Float16* bp = hT + (size_t)(n0 + nt * 16 + l) * N_NODES + j0 + kb + halfw * 8;
        h8 b0 = *(const h8*)bp;
        h8 b1 = *(const h8*)(bp + 16);
        v16h Bm = __builtin_shufflevector(b0, b1, 0,1,2,3,4,5,6,7,8,9,10,11,12,13,14,15);
        if (nt == 0)
          acc0 = __builtin_amdgcn_wmma_f32_16x16x32_f16(false, Am, false, Bm,
                                                        (short)0, acc0, false, false);
        else
          acc1 = __builtin_amdgcn_wmma_f32_16x16x32_f16(false, Am, false, Bm,
                                                        (short)0, acc1, false, false);
      }
    }
  }

  // row-sum reduce: row r partials live at tid = r*16 .. r*16+15
  red[tid] = psum;
  __syncthreads();
  if (tid < 16) {
    float s = 0.f;
#pragma unroll
    for (int k = 0; k < 16; ++k) s += red[tid * 16 + k];
    sLinv[tid] = s > 0.f ? 1.0f / s : 0.f;
  }
  __syncthreads();

#pragma unroll
  for (int g = 0; g < 8; ++g) {
    int rr = halfw * 8 + g;              // C/D layout row
    float li = sLinv[rr];
    float v0 = acc0[g] * li;
    float v1 = acc1[g] * li;
    out[(size_t)(r0 + rr) * FDIM + n0 + l]      = v0 > 0.f ? v0 : 0.f;
    out[(size_t)(r0 + rr) * FDIM + n0 + 16 + l] = v1 > 0.f ? v1 : 0.f;
  }
}

// =====================================================================
// Host launcher. Workspace layout: hT (4 MB f16) | src | dst | maxdst.
// h (fp32, 8 MB) is staged in d_out and overwritten by the final kernel.
// =====================================================================
extern "C" void kernel_launch(void* const* d_in, const int* in_sizes, int n_in,
                              void* d_out, int out_size, void* d_ws, size_t ws_size,
                              hipStream_t stream) {
  const float* input = (const float*)d_in[0];
  const int*   adj   = (const int*)d_in[1];
  const float* W     = (const float*)d_in[2];
  const float* avec  = (const float*)d_in[3];
  float* out = (float*)d_out;

  char* ws = (char*)d_ws;
  _Float16* hT  = (_Float16*)ws;                                   // 4 MB
  float* src    = (float*)(ws + (size_t)N_NODES * FDIM * 2);       // 32 KB
  float* dst    = src + N_NODES;                                   // 32 KB
  float* maxdst = dst + N_NODES;                                   // 4 B

  gat_h_gemm <<<N_NODES / 16, 512, 0, stream>>>(input, W, out /*h staged here*/, hT);
  gat_src_dst<<<N_NODES / 8,  256, 0, stream>>>(out, avec, src, dst);
  gat_maxdst <<<1,            256, 0, stream>>>(dst, maxdst);
  gat_attn   <<<N_NODES / 16, 256, 0, stream>>>(adj, hT, src, dst, maxdst, out);
}
